// Net2_35158602285514
// MI455X (gfx1250) — matrix operand order repaired and checked by reading, not compiled, not run on hardware
//
#include <hip/hip_runtime.h>

typedef __attribute__((ext_vector_type(16))) _Float16 v16h;
typedef __attribute__((ext_vector_type(8)))  float    v8f;

__global__ void zero_f32(float* __restrict__ p, long long n) {
  long long i = (long long)blockIdx.x * blockDim.x + threadIdx.x;
  if (i < n) p[i] = 0.0f;
}

__global__ void degree_k(const int* __restrict__ row, float* __restrict__ deg, int E) {
  int e = blockIdx.x * blockDim.x + threadIdx.x;
  if (e < E) atomicAdd(&deg[row[e]], 1.0f);
}

__global__ void dinv_k(const float* __restrict__ deg, float* __restrict__ dinv, int N) {
  int i = blockIdx.x * blockDim.x + threadIdx.x;
  if (i < N) {
    float d = deg[i];
    dinv[i] = (d > 0.0f) ? rsqrtf(d) : 0.0f;
  }
}

// W[K,64] f32 row-major  ->  Wt[64,K] f16 (transposed), so a B fragment
// (16 consecutive K for one output column) is a contiguous 32-byte load.
__global__ void convw_k(const float* __restrict__ W, _Float16* __restrict__ Wt, int K) {
  int t = blockIdx.x * blockDim.x + threadIdx.x;
  if (t >= K * 64) return;
  int k = t >> 6, c = t & 63;
  Wt[(size_t)c * K + k] = (_Float16)W[t];
}

// out[16 rows x 64 cols] = (RELU ? relu(in) : in)[rows, K] @ W[K,64] + bias
// Block = 128 threads (4 waves); wave w computes output columns [16w, 16w+16).
// Fully unrolled K loop -> K/32 v_wmma_f32_16x16x32_f16 per wave.
template <int K, bool RELU>
__global__ __launch_bounds__(128) void linear_wmma(
    const float* __restrict__ in, const _Float16* __restrict__ Wt,
    const float* __restrict__ bias, float* __restrict__ out, int N) {
  const int wave = threadIdx.x >> 5;
  const int lane = threadIdx.x & 31;
  const int n    = lane & 15;        // column within 16-wide tile / A row
  const int hi   = lane >> 4;        // lane half
  const int klo  = hi << 4;          // A-fragment K offset (contiguous halves)
  const int row0 = blockIdx.x << 4;
  const int rA   = min(row0 + n, N - 1);   // clamp keeps EXEC all-ones
  const float*    inrow = in + (size_t)rA * K;
  const int       col   = (wave << 4) + n;
  const _Float16* wcol  = Wt + (size_t)col * K;

  v8f c = {};
#pragma unroll
  for (int k0 = 0; k0 < K; k0 += 32) {
    // A: 16x32 f16, row M = lane%16; lanes 0-15 hold K=0..15, lanes 16-31 hold K=16..31
    const float4 p0 = *(const float4*)(inrow + k0 + klo);
    const float4 p1 = *(const float4*)(inrow + k0 + klo + 4);
    const float4 p2 = *(const float4*)(inrow + k0 + klo + 8);
    const float4 p3 = *(const float4*)(inrow + k0 + klo + 12);
    float f[16] = {p0.x, p0.y, p0.z, p0.w, p1.x, p1.y, p1.z, p1.w,
                   p2.x, p2.y, p2.z, p2.w, p3.x, p3.y, p3.z, p3.w};
    v16h a;
#pragma unroll
    for (int j = 0; j < 16; ++j) {
      float v = RELU ? fmaxf(f[j], 0.0f) : f[j];
      a[j] = (_Float16)v;
    }
    // B: 32x16 f16, col N = lane%16; lanes 0-15 hold K=0..15, lanes 16-31 hold K=16..31
    v16h b = *(const v16h*)(wcol + k0 + (hi << 4));   // contiguous 32B
    c = __builtin_amdgcn_wmma_f32_16x16x32_f16(false, a, false, b,
                                               (short)0, c, false, false);
  }
  // C/D layout: VGPR r -> row M = r + 8*hi, col = lane%16
  const float bv = bias[col];
#pragma unroll
  for (int r = 0; r < 8; ++r) {
    int m = row0 + r + (hi << 3);
    if (m < N) out[(size_t)m * 64 + col] = c[r] + bv;
  }
}

// 16 threads per edge, float4 per thread:
// out[row, 4f..4f+4) += dinv[row]*dinv[col] * h[col, 4f..4f+4)
__global__ void aggregate_k(const float* __restrict__ h, const int* __restrict__ row,
                            const int* __restrict__ col, const float* __restrict__ dinv,
                            float* __restrict__ out, int E) {
  long long t = (long long)blockIdx.x * blockDim.x + threadIdx.x;
  int e  = (int)(t >> 4);
  int f4 = (int)(t & 15) << 2;
  if (e >= E) return;
  int r = row[e], s = col[e];
  float norm = dinv[r] * dinv[s];
  const float4 hv = *(const float4*)(h + (size_t)s * 64 + f4);
  float* o = out + (size_t)r * 64 + f4;
  atomicAdd(o + 0, norm * hv.x);
  atomicAdd(o + 1, norm * hv.y);
  atomicAdd(o + 2, norm * hv.z);
  atomicAdd(o + 3, norm * hv.w);
}

// One wave32 per edge: out[e] = dot(z[src], z[dst]) over 64 dims (float2 per lane)
__global__ void decode_k(const float* __restrict__ z, const int* __restrict__ pos,
                         const int* __restrict__ neg, int Ep, int En,
                         float* __restrict__ out) {
  long long t = (long long)blockIdx.x * blockDim.x + threadIdx.x;
  int e    = (int)(t >> 5);
  int lane = (int)(t & 31);
  if (e >= Ep + En) return;
  int a, b;
  if (e < Ep) { a = pos[e]; b = pos[Ep + e]; }
  else        { int e2 = e - Ep; a = neg[e2]; b = neg[En + e2]; }
  const float2 pa = *(const float2*)(z + (size_t)a * 64 + 2 * lane);
  const float2 pb = *(const float2*)(z + (size_t)b * 64 + 2 * lane);
  float s = pa.x * pb.x + pa.y * pb.y;
#pragma unroll
  for (int off = 16; off > 0; off >>= 1) s += __shfl_down(s, off, 32);
  if (lane == 0) out[e] = s;
}

extern "C" void kernel_launch(void* const* d_in, const int* in_sizes, int n_in,
                              void* d_out, int out_size, void* d_ws, size_t ws_size,
                              hipStream_t stream) {
  const float* x   = (const float*)d_in[0];
  const int*   pos = (const int*)d_in[1];
  const int*   neg = (const int*)d_in[2];
  const float* W1  = (const float*)d_in[3];
  const float* b1  = (const float*)d_in[4];
  const float* W2  = (const float*)d_in[5];
  const float* b2  = (const float*)d_in[6];
  float* out = (float*)d_out;

  const int N  = in_sizes[0] / 256;   // 100000
  const int Ep = in_sizes[1] / 2;     // 1.6M
  const int En = in_sizes[2] / 2;     // 1.6M

  float* deg  = (float*)d_ws;                 // [N]
  float* dinv = deg + N;                      // [N]
  float* h    = dinv + N;                     // [N,64] linear output (reused both layers)
  float* agg  = h + (size_t)N * 64;           // [N,64] layer-1 aggregation
  float* z    = agg + (size_t)N * 64;         // [N,64] final embeddings
  _Float16* Wt1 = (_Float16*)(z + (size_t)N * 64); // [64,256] f16 transposed W1
  _Float16* Wt2 = Wt1 + 256 * 64;                  // [64,64]  f16 transposed W2

  const int T = 256;
  const long long n64 = (long long)N * 64;

  zero_f32<<<(unsigned)((N + T - 1) / T),   T, 0, stream>>>(deg, N);
  zero_f32<<<(unsigned)((n64 + T - 1) / T), T, 0, stream>>>(agg, n64);
  zero_f32<<<(unsigned)((n64 + T - 1) / T), T, 0, stream>>>(z, n64);

  convw_k<<<(256 * 64 + T - 1) / T, T, 0, stream>>>(W1, Wt1, 256);
  convw_k<<<(64 * 64 + T - 1) / T,  T, 0, stream>>>(W2, Wt2, 64);

  degree_k<<<(Ep + T - 1) / T, T, 0, stream>>>(pos, deg, Ep);
  dinv_k<<<(N + T - 1) / T, T, 0, stream>>>(deg, dinv, N);

  const int nblk = (N + 15) / 16;
  // Layer 1 linear: h = x @ W1 + b1   (WMMA f16->f32, K=256 fully unrolled)
  linear_wmma<256, false><<<nblk, 128, 0, stream>>>(x, Wt1, b1, h, N);

  const long long at = (long long)Ep * 16;
  // Layer 1 aggregation: agg[row] += norm * h[col]
  aggregate_k<<<(unsigned)((at + T - 1) / T), T, 0, stream>>>(h, pos, pos + Ep, dinv, agg, Ep);

  // Layer 2 linear (ReLU fused on load): h = relu(agg) @ W2 + b2  (K=64)
  linear_wmma<64, true><<<nblk, 128, 0, stream>>>(agg, Wt2, b2, h, N);

  // Layer 2 aggregation: z[row] += norm * h[col]
  aggregate_k<<<(unsigned)((at + T - 1) / T), T, 0, stream>>>(h, pos, pos + Ep, dinv, z, Ep);

  // Decode: dot(z[src], z[dst]) over pos then neg edges
  const long long dt = (long long)(Ep + En) * 32;
  decode_k<<<(unsigned)((dt + T - 1) / T), T, 0, stream>>>(z, pos, neg, Ep, En, out);
}
